// update_v_88347477279414
// MI455X (gfx1250) — matrix operand order repaired and checked by reading, not compiled, not run on hardware
//
#include <hip/hip_runtime.h>
#include <hip/hip_bf16.h>
#include <stdint.h>

// ---------------- sizes ----------------
#define NND   24000
#define EBN   384000
#define EMSG  192000

typedef __attribute__((ext_vector_type(16))) __bf16   bf16x16;
typedef __attribute__((ext_vector_type(8)))  float    f32x8;
typedef __attribute__((ext_vector_type(4)))  float    f32x4;
typedef __attribute__((ext_vector_type(4)))  uint32_t u32x4;

union FragAB { bf16x16 v; uint32_t u[8]; u32x4 q[2]; };

__device__ __forceinline__ uint32_t pk2bf(float a, float b) {
  uint32_t ua = __float_as_uint(a), ub = __float_as_uint(b);
  return ((ub + 0x8000u) & 0xFFFF0000u) | (((ua + 0x8000u) >> 16) & 0xFFFFu);
}
__device__ __forceinline__ uint16_t f2bf(float a) {
  return (uint16_t)((__float_as_uint(a) + 0x8000u) >> 16);
}
__device__ __forceinline__ f32x8 wmma_bf16(const FragAB& a, const FragAB& b, f32x8 c) {
  return __builtin_amdgcn_wmma_f32_16x16x32_bf16(false, a.v, false, b.v, (short)0, c, false, false);
}
__device__ __forceinline__ uint32_t enc_ord(float f) {
  uint32_t b = __float_as_uint(f);
  return (b & 0x80000000u) ? ~b : (b | 0x80000000u);
}
__device__ __forceinline__ float dec_ord(uint32_t u) {
  uint32_t b = (u & 0x80000000u) ? (u ^ 0x80000000u) : ~u;
  return __uint_as_float(b);
}

// load one B fragment (16 consecutive bf16 per lane) as 2x b128 from LDS
__device__ __forceinline__ void ldfrag(FragAB& d, const char* bbase, int byteOff) {
  const u32x4* p = (const u32x4*)(bbase + byteOff);
  d.q[0] = p[0];
  d.q[1] = p[1];
}

// build A fragment from 16 fp32 values at ak (16B aligned), scaled by s.
// layout: lane<16 -> K {0..7,16..23}, lane>=16 -> K {8..15,24..31} (ksel folded into ak)
__device__ __forceinline__ void buildA(FragAB& af, const float* ak, float s) {
  f32x4 a0 = *(const f32x4*)(ak);
  f32x4 a1 = *(const f32x4*)(ak + 4);
  f32x4 a2 = *(const f32x4*)(ak + 16);
  f32x4 a3 = *(const f32x4*)(ak + 20);
  af.u[0] = pk2bf(s * a0[0], s * a0[1]);
  af.u[1] = pk2bf(s * a0[2], s * a0[3]);
  af.u[2] = pk2bf(s * a1[0], s * a1[1]);
  af.u[3] = pk2bf(s * a1[2], s * a1[3]);
  af.u[4] = pk2bf(s * a2[0], s * a2[1]);
  af.u[5] = pk2bf(s * a2[2], s * a2[3]);
  af.u[6] = pk2bf(s * a3[0], s * a3[1]);
  af.u[7] = pk2bf(s * a3[2], s * a3[3]);
}

// vectorized global->LDS slab copy (ndw dwords, multiple of 4) + L2 prefetch
template<int BLK>
__device__ __forceinline__ void copy_slab(uint32_t* dst, const uint32_t* src, int ndw,
                                          int tid, const uint32_t* pref) {
  const u32x4* s4 = (const u32x4*)src;
  u32x4* d4 = (u32x4*)dst;
  int nv = ndw >> 2;
  for (int i = tid; i < nv; i += BLK) d4[i] = s4[i];
  if (pref) __builtin_prefetch((const char*)pref + (size_t)tid * 128, 0, 1);
}

// ======================================================================
// Generic row-tile WMMA GEMM: out[N, NT*16] = act(A[N,kact] @ W + bias)
// block = 128 threads (4 waves), wave owns a 16-row tile; kact == kch*32.
// NT must be even. Fragments issued in pairs to overlap LDS latency.
// ======================================================================
template<int NT, int ACT>
__global__ void __launch_bounds__(128) gemm_bf16_kernel(
    const float* __restrict__ A, int kact, int kch,
    const uint16_t* __restrict__ Wsw, const float* __restrict__ bias,
    float* __restrict__ out, int nrows)
{
  extern __shared__ char smem[];
  const int kpad = kch * 32, pitch = kpad + 4;
  float*    sA  = (float*)smem;
  uint32_t* sB0 = (uint32_t*)(smem + (size_t)64 * pitch * 4);
  uint32_t* sB1 = sB0 + NT * 256;
  const int tid = threadIdx.x, wave = tid >> 5, lane = tid & 31;
  const int rowBase = blockIdx.x * 64;

  {
    const int kv4 = kact >> 2;
    const f32x4* g4 = (const f32x4*)(A + (size_t)rowBase * kact);
    f32x4* s4 = (f32x4*)sA;
    const int p4 = pitch >> 2;
    for (int idx = tid; idx < 64 * kv4; idx += 128) {
      int r = idx / kv4, k4 = idx - r * kv4;
      s4[r * p4 + k4] = g4[idx];
    }
  }
  const uint32_t* gW = (const uint32_t*)Wsw;
  copy_slab<128>(sB0, gW, NT * 256, tid, gW + NT * 256);
  __syncthreads();

  f32x8 acc[NT] = {};
  const float* arow = sA + (wave * 16 + (lane & 15)) * pitch;
  const int ksel = (lane < 16) ? 0 : 8;
  const char* bb0 = (const char*)sB0 + lane * 32;
  const char* bb1 = (const char*)sB1 + lane * 32;

  for (int c = 0; c < kch; ++c) {
    if (c + 1 < kch)
      copy_slab<128>((c & 1) ? sB0 : sB1, gW + (size_t)(c + 1) * NT * 256, NT * 256, tid,
                     (c + 2 < kch) ? gW + (size_t)(c + 2) * NT * 256 : nullptr);
    FragAB af;
    buildA(af, arow + c * 32 + ksel, 1.0f);
    const char* bb = (c & 1) ? bb1 : bb0;
#pragma unroll
    for (int nt = 0; nt < NT; nt += 2) {
      FragAB fx, fy;
      ldfrag(fx, bb, nt * 1024);
      ldfrag(fy, bb, (nt + 1) * 1024);
      acc[nt]     = wmma_bf16(af, fx, acc[nt]);
      acc[nt + 1] = wmma_bf16(af, fy, acc[nt + 1]);
    }
    __syncthreads();
  }

  const int mrow = rowBase + wave * 16 + ((lane < 16) ? 0 : 8);
#pragma unroll
  for (int nt = 0; nt < NT; ++nt) {
    int col = nt * 16 + (lane & 15);
    float bv = bias ? bias[col] : 0.f;
#pragma unroll
    for (int r = 0; r < 8; ++r) {
      int row = mrow + r;
      float vv = acc[nt][r] + bv;
      if (ACT == 1) vv = fmaxf(vv, 0.f);
      else if (ACT == 2) vv = vv / (1.f + __expf(-vv));
      if (row < nrows) out[(size_t)row * (NT * 16) + col] = vv;
    }
  }
}

// ======================================================================
// Bilinear stage 1: z[b,k] = sum_ij v1[b,i]*x[b,j]*Bcat[ij,k] + bzcat[k]
// Dual-A: block = 64 threads (2 waves), each wave owns 32 rows (2 A tiles),
// N split across gridDim.y (6 n-tiles each). Each B fragment feeds 2 WMMAs.
// ======================================================================
__global__ void __launch_bounds__(64) bilinear1_kernel(
    const float* __restrict__ V, const float* __restrict__ X,
    const uint16_t* __restrict__ Bsw, const float* __restrict__ bias,
    float* __restrict__ out, int nrows)
{
  extern __shared__ char smem[];
  const int PV = 132, PX = 68;
  float*    sV  = (float*)smem;
  float*    sX  = sV + 64 * PV;
  uint32_t* sB0 = (uint32_t*)(sX + 64 * PX);
  uint32_t* sB1 = sB0 + 6 * 256;
  const int tid = threadIdx.x, wave = tid >> 5, lane = tid & 31;
  const int rowBase = blockIdx.x * 64;
  const int ntBase = blockIdx.y * 6;

  {
    const f32x4* g4 = (const f32x4*)(V + (size_t)rowBase * 128);
    f32x4* s4 = (f32x4*)sV;
    for (int idx = tid; idx < 64 * 32; idx += 64) {
      int r = idx >> 5, k4 = idx & 31;
      s4[r * (PV >> 2) + k4] = g4[idx];
    }
  }
  {
    const f32x4* g4 = (const f32x4*)(X + (size_t)rowBase * 64);
    f32x4* s4 = (f32x4*)sX;
    for (int idx = tid; idx < 64 * 16; idx += 64) {
      int r = idx >> 4, k4 = idx & 15;
      s4[r * (PX >> 2) + k4] = g4[idx];
    }
  }
  const uint32_t* gW = (const uint32_t*)Bsw + (size_t)ntBase * 256;
  copy_slab<64>(sB0, gW, 6 * 256, tid, gW + 12 * 256);
  __syncthreads();

  f32x8 acc0[6] = {}, acc1[6] = {};
  const float* vrow0 = sV + (wave * 32 + (lane & 15)) * PV;
  const float* xrow0 = sX + (wave * 32 + (lane & 15)) * PX;
  const float* vrow1 = vrow0 + 16 * PV;
  const float* xrow1 = xrow0 + 16 * PX;
  const int ksel = (lane < 16) ? 0 : 8;
  const char* bb0 = (const char*)sB0 + lane * 32;
  const char* bb1 = (const char*)sB1 + lane * 32;

  for (int c = 0; c < 256; ++c) {
    if (c + 1 < 256)
      copy_slab<64>((c & 1) ? sB0 : sB1, gW + (size_t)(c + 1) * 12 * 256, 6 * 256, tid,
                    (c + 2 < 256) ? gW + (size_t)(c + 2) * 12 * 256 : nullptr);
    const int i = c >> 1, joff = ((c & 1) << 5) + ksel;
    FragAB a0, a1;
    buildA(a0, xrow0 + joff, vrow0[i]);
    buildA(a1, xrow1 + joff, vrow1[i]);
    const char* bb = (c & 1) ? bb1 : bb0;
#pragma unroll
    for (int nt = 0; nt < 6; nt += 2) {
      FragAB fx, fy;
      ldfrag(fx, bb, nt * 1024);
      ldfrag(fy, bb, (nt + 1) * 1024);
      acc0[nt]     = wmma_bf16(a0, fx, acc0[nt]);
      acc1[nt]     = wmma_bf16(a1, fx, acc1[nt]);
      acc0[nt + 1] = wmma_bf16(a0, fy, acc0[nt + 1]);
      acc1[nt + 1] = wmma_bf16(a1, fy, acc1[nt + 1]);
    }
    __syncthreads();
  }

  const int mrow = rowBase + wave * 32 + ((lane < 16) ? 0 : 8);
#pragma unroll
  for (int nt = 0; nt < 6; ++nt) {
    int col = (ntBase + nt) * 16 + (lane & 15);
    float bv = bias[col];
#pragma unroll
    for (int r = 0; r < 8; ++r) {
      int row0 = mrow + r, row1 = row0 + 16;
      if (row0 < nrows) out[(size_t)row0 * 192 + col] = acc0[nt][r] + bv;
      if (row1 < nrows) out[(size_t)row1 * 192 + col] = acc1[nt][r] + bv;
    }
  }
}

// ======================================================================
// Bilinear stage 2: out192 = relu( [o1 (x) o2 | o1 | o2] @ W2 + be1_eff )
// Dual-A + N-split (see bilinear1). 256 product chunks + 6 direct chunks.
// ======================================================================
__global__ void __launch_bounds__(64) bilinear2_kernel(
    const float* __restrict__ L, const uint16_t* __restrict__ Wsw,
    const float* __restrict__ bias, float* __restrict__ out, int nrows)
{
  extern __shared__ char smem[];
  const int PL = 196;
  float*    sL  = (float*)smem;
  uint32_t* sB0 = (uint32_t*)(sL + 64 * PL);
  uint32_t* sB1 = sB0 + 6 * 256;
  const int tid = threadIdx.x, wave = tid >> 5, lane = tid & 31;
  const int rowBase = blockIdx.x * 64;
  const int ntBase = blockIdx.y * 6;

  {
    const f32x4* g4 = (const f32x4*)(L + (size_t)rowBase * 192);
    f32x4* s4 = (f32x4*)sL;
    for (int idx = tid; idx < 64 * 48; idx += 64) {
      int r = idx / 48, k4 = idx - r * 48;
      s4[r * (PL >> 2) + k4] = g4[idx];
    }
  }
  const uint32_t* gW = (const uint32_t*)Wsw + (size_t)ntBase * 256;
  copy_slab<64>(sB0, gW, 6 * 256, tid, gW + 12 * 256);
  __syncthreads();

  f32x8 acc0[6] = {}, acc1[6] = {};
  const float* lrow0 = sL + (wave * 32 + (lane & 15)) * PL;
  const float* lrow1 = lrow0 + 16 * PL;
  const int ksel = (lane < 16) ? 0 : 8;
  const char* bb0 = (const char*)sB0 + lane * 32;
  const char* bb1 = (const char*)sB1 + lane * 32;

  for (int c = 0; c < 262; ++c) {
    if (c + 1 < 262)
      copy_slab<64>((c & 1) ? sB0 : sB1, gW + (size_t)(c + 1) * 12 * 256, 6 * 256, tid,
                    (c + 2 < 262) ? gW + (size_t)(c + 2) * 12 * 256 : nullptr);
    FragAB a0, a1;
    if (c < 256) {  // outer-product region: o1[i] * o2[j]
      const int i = c >> 1, joff = 128 + ((c & 1) << 5) + ksel;
      buildA(a0, lrow0 + joff, lrow0[i]);
      buildA(a1, lrow1 + joff, lrow1[i]);
    } else {        // direct linear region: [o1 | o2]
      const int doff = (c - 256) * 32 + ksel;
      buildA(a0, lrow0 + doff, 1.0f);
      buildA(a1, lrow1 + doff, 1.0f);
    }
    const char* bb = (c & 1) ? bb1 : bb0;
#pragma unroll
    for (int nt = 0; nt < 6; nt += 2) {
      FragAB fx, fy;
      ldfrag(fx, bb, nt * 1024);
      ldfrag(fy, bb, (nt + 1) * 1024);
      acc0[nt]     = wmma_bf16(a0, fx, acc0[nt]);
      acc1[nt]     = wmma_bf16(a1, fx, acc1[nt]);
      acc0[nt + 1] = wmma_bf16(a0, fy, acc0[nt + 1]);
      acc1[nt + 1] = wmma_bf16(a1, fy, acc1[nt + 1]);
    }
    __syncthreads();
  }

  const int mrow = rowBase + wave * 32 + ((lane < 16) ? 0 : 8);
#pragma unroll
  for (int nt = 0; nt < 6; ++nt) {
    int col = (ntBase + nt) * 16 + (lane & 15);
    float bv = bias[col];
#pragma unroll
    for (int r = 0; r < 8; ++r) {
      int row0 = mrow + r, row1 = row0 + 16;
      if (row0 < nrows) out[(size_t)row0 * 192 + col] = fmaxf(acc0[nt][r] + bv, 0.f);
      if (row1 < nrows) out[(size_t)row1 * 192 + col] = fmaxf(acc1[nt][r] + bv, 0.f);
    }
  }
}

// ---------------- scatter / GAT / elementwise ----------------
__global__ void scatter_rows_kernel(const float* __restrict__ e2,
                                    const int* __restrict__ idx,
                                    float* __restrict__ v, int emsg) {
  long t = (long)blockIdx.x * blockDim.x + threadIdx.x;
  if (t >= (long)emsg * 128) return;
  int e = (int)(t >> 7), c = (int)(t & 127);
  atomicAdd(&v[(size_t)idx[e] * 128 + c], e2[t]);
}

__global__ void gat_node_kernel(const float* __restrict__ xin, const float* __restrict__ W,
                                const float* __restrict__ asW, const float* __restrict__ adW,
                                float* __restrict__ h, float* __restrict__ asrc,
                                float* __restrict__ adst) {
  __shared__ float xr[64], r1[64], r2[64];
  int n = blockIdx.x, c = threadIdx.x;
  xr[c] = xin[(size_t)n * 64 + c];
  __syncthreads();
  float s = 0.f;
  for (int k = 0; k < 64; ++k) s += xr[k] * W[k * 64 + c];
  h[(size_t)n * 64 + c] = s;
  r1[c] = s * asW[c]; r2[c] = s * adW[c];
  __syncthreads();
  for (int off = 32; off; off >>= 1) {
    if (c < off) { r1[c] += r1[c + off]; r2[c] += r2[c + off]; }
    __syncthreads();
  }
  if (c == 0) { asrc[n] = r1[0]; adst[n] = r2[0]; }
}

__global__ void init_gat_out_kernel(float* __restrict__ xo, const float* __restrict__ bias,
                                    uint32_t* __restrict__ amax, float* __restrict__ den, int n) {
  long t = (long)blockIdx.x * blockDim.x + threadIdx.x;
  if (t >= (long)n * 64) return;
  int node = (int)(t >> 6), c = (int)(t & 63);
  xo[t] = bias[c];
  if (c == 0) { amax[node] = 0x007FFFFFu; den[node] = 0.f; }  // enc(-inf)
}

__global__ void we8_kernel(const float* __restrict__ We, const float* __restrict__ ae,
                           float* __restrict__ we8) {
  int d = threadIdx.x; if (d >= 8) return;
  float s = 0.f;
  for (int c = 0; c < 64; ++c) s += We[d * 64 + c] * ae[c];
  we8[d] = s;
}

__global__ void edge1_kernel(const int* __restrict__ ei, const float* __restrict__ attr,
                             const float* __restrict__ asrc, const float* __restrict__ adst,
                             const float* __restrict__ we8, float* __restrict__ aE,
                             uint32_t* __restrict__ amax, int eb) {
  int e = blockIdx.x * blockDim.x + threadIdx.x;
  if (e >= eb) return;
  int sn = ei[e], dn = ei[eb + e];
  float a = asrc[sn] + adst[dn];
  const float* at = attr + (size_t)e * 8;
#pragma unroll
  for (int k = 0; k < 8; ++k) a += at[k] * we8[k];
  a = a > 0.f ? a : 0.2f * a;
  aE[e] = a;
  atomicMax(&amax[dn], enc_ord(a));
}

__global__ void edge2_kernel(const int* __restrict__ ei, const float* __restrict__ aE,
                             const uint32_t* __restrict__ amax, float* __restrict__ exE,
                             float* __restrict__ den, int eb) {
  int e = blockIdx.x * blockDim.x + threadIdx.x;
  if (e >= eb) return;
  int dn = ei[eb + e];
  float mx = dec_ord(amax[dn]);
  if (!__builtin_isfinite(mx)) mx = 0.f;
  float ex = __expf(aE[e] - mx);
  exE[e] = ex;
  atomicAdd(&den[dn], ex);
}

__global__ void edge3_kernel(const int* __restrict__ ei, const float* __restrict__ exE,
                             const float* __restrict__ den, const float* __restrict__ h,
                             float* __restrict__ xo, int eb) {
  long t = (long)blockIdx.x * blockDim.x + threadIdx.x;
  if (t >= (long)eb * 64) return;
  int e = (int)(t >> 6), c = (int)(t & 63);
  int sn = ei[e], dn = ei[eb + e];
  float w = exE[e] / (den[dn] + 1e-16f);
  atomicAdd(&xo[(size_t)dn * 64 + c], h[(size_t)sn * 64 + c] * w);
}

__global__ void relu_kernel(float* __restrict__ x, long n) {
  long t = (long)blockIdx.x * blockDim.x + threadIdx.x;
  if (t < n) x[t] = fmaxf(x[t], 0.f);
}

__global__ void gate_kernel(const float* __restrict__ z, int zoff,
                            const float* __restrict__ h, float* __restrict__ g,
                            int n, int cols) {
  long t = (long)blockIdx.x * blockDim.x + threadIdx.x;
  if (t >= (long)n * cols) return;
  int row = (int)(t / cols), j = (int)(t - (long)row * cols);
  float zz = z[(size_t)row * 192 + zoff + j];
  g[t] = h[t] * (1.f / (1.f + __expf(-zz)));
}

__global__ void concat2_kernel(const float* __restrict__ a, int ca,
                               const float* __restrict__ b, int cb,
                               float* __restrict__ o, int n) {
  int c = ca + cb;
  long t = (long)blockIdx.x * blockDim.x + threadIdx.x;
  if (t >= (long)n * c) return;
  int row = (int)(t / c), j = (int)(t - (long)row * c);
  o[t] = (j < ca) ? a[(size_t)row * ca + j] : b[(size_t)row * cb + (j - ca)];
}

__global__ void concat3_kernel(const float* __restrict__ a, int ca,
                               const float* __restrict__ b, int cb,
                               const float* __restrict__ cc, int ccn,
                               float* __restrict__ o, int n) {
  int c = ca + cb + ccn;
  long t = (long)blockIdx.x * blockDim.x + threadIdx.x;
  if (t >= (long)n * c) return;
  int row = (int)(t / c), j = (int)(t - (long)row * c);
  float vv;
  if (j < ca) vv = a[(size_t)row * ca + j];
  else if (j < ca + cb) vv = b[(size_t)row * cb + (j - ca)];
  else vv = cc[(size_t)row * ccn + (j - ca - cb)];
  o[t] = vv;
}

// ---------------- weight prep ----------------
__global__ void make_bcat_kernel(const float* __restrict__ B1, const float* __restrict__ B2,
                                 float* __restrict__ o) {
  long t = (long)blockIdx.x * blockDim.x + threadIdx.x;
  if (t >= 8192L * 192) return;
  int ij = (int)(t / 192), k = (int)(t - (long)ij * 192);
  o[t] = (k < 128) ? B1[(size_t)k * 8192 + ij] : B2[(size_t)(k - 128) * 8192 + ij];
}

__global__ void make_w2_kernel(const float* __restrict__ We1, float* __restrict__ o) {
  long t = (long)blockIdx.x * blockDim.x + threadIdx.x;
  if (t >= 8384L * 192) return;
  int r = (int)(t / 192), k = (int)(t - (long)r * 192);
  int src;
  if (r < 8192)      { int p = r >> 6, q = r & 63; src = p * 65 + q; }
  else if (r < 8320) { int p = r - 8192;           src = p * 65 + 64; }
  else               { int q = r - 8320;           src = 8320 + q; }
  o[t] = We1[(size_t)src * 192 + k];
}

__global__ void make_we2c_kernel(const float* __restrict__ We2, float* __restrict__ o) {
  long t = (long)blockIdx.x * blockDim.x + threadIdx.x;
  if (t >= 384L * 192) return;
  int r = (int)(t / 192), k = (int)(t - (long)r * 192);
  int src = (r < 320) ? r : r + 1;     // skip constant rows 320 / 385
  o[t] = We2[(size_t)src * 192 + k];
}

__global__ void bias_cat_kernel(const float* __restrict__ b1, int c1,
                                const float* __restrict__ b2, int c2,
                                float* __restrict__ o) {
  int t = threadIdx.x; if (t >= c1 + c2) return;
  o[t] = (t < c1) ? b1[t] : b2[t - c1];
}
__global__ void be1eff_kernel(const float* __restrict__ be1, const float* __restrict__ We1,
                              float* __restrict__ o) {
  int k = threadIdx.x; if (k >= 192) return;
  o[k] = be1[k] + We1[(size_t)8384 * 192 + k];
}
__global__ void be2eff_kernel(const float* __restrict__ be2, const float* __restrict__ We2,
                              float* __restrict__ o) {
  int k = threadIdx.x; if (k >= 192) return;
  o[k] = be2[k] + We2[(size_t)320 * 192 + k] + We2[(size_t)385 * 192 + k];
}

// swizzle row-major fp32 W[kact, ntiles*16] -> per-fragment bf16 layout
__global__ void swizzle_kernel(const float* __restrict__ W, uint16_t* __restrict__ o,
                               int kact, int kch, int ntiles) {
  long total = (long)kch * ntiles * 512;
  long t = (long)blockIdx.x * blockDim.x + threadIdx.x;
  if (t >= total) return;
  int s    = (int)(t & 15);
  int lane = (int)((t >> 4) & 31);
  long rest = t >> 9;
  int nt = (int)(rest % ntiles);
  int c  = (int)(rest / ntiles);
  int K   = c * 32 + ((s < 8) ? s : 8 + s) + ((lane < 16) ? 0 : 8);
  int col = nt * 16 + (lane & 15);
  float vv = (K < kact) ? W[(size_t)K * (ntiles * 16) + col] : 0.f;
  o[t] = f2bf(vv);
}

__global__ void wout_kernel(const float* __restrict__ v, const float* __restrict__ W,
                            float* __restrict__ o, int n) {
  int t = blockIdx.x * blockDim.x + threadIdx.x;
  if (t >= n) return;
  const float* r = v + (size_t)t * 256;
  float s = 0.f;
  for (int k = 0; k < 256; ++k) s += r[k] * W[k];
  o[t] = s;
}

// ======================================================================
extern "C" void kernel_launch(void* const* d_in, const int* in_sizes, int n_in,
                              void* d_out, int out_size, void* d_ws, size_t ws_size,
                              hipStream_t stream) {
  (void)in_sizes; (void)n_in; (void)out_size; (void)ws_size;
  // params in setup_inputs() dict (insertion) order
  const float* gW    = (const float*)d_in[0];
  const float* gbias = (const float*)d_in[1];
  const float* gas   = (const float*)d_in[2];
  const float* gad   = (const float*)d_in[3];
  const float* gWe   = (const float*)d_in[4];
  const float* gae   = (const float*)d_in[5];
  const float* Wh1 = (const float*)d_in[6];  const float* bh1 = (const float*)d_in[7];
  const float* B1  = (const float*)d_in[8];  const float* bz1 = (const float*)d_in[9];
  const float* Wo1 = (const float*)d_in[10]; const float* bo1 = (const float*)d_in[11];
  const float* Wh2 = (const float*)d_in[12]; const float* bh2 = (const float*)d_in[13];
  const float* B2  = (const float*)d_in[14]; const float* bz2 = (const float*)d_in[15];
  const float* Wo2 = (const float*)d_in[16]; const float* bo2 = (const float*)d_in[17];
  const float* We1 = (const float*)d_in[18]; const float* be1 = (const float*)d_in[19];
  const float* We2 = (const float*)d_in[20]; const float* be2 = (const float*)d_in[21];
  const float* Wd  = (const float*)d_in[22]; const float* bd  = (const float*)d_in[23];
  const float* Wu  = (const float*)d_in[24]; const float* bu  = (const float*)d_in[25];
  const float* L0W = (const float*)d_in[26]; const float* L0b = (const float*)d_in[27];
  const float* L1W = (const float*)d_in[28]; const float* L1b = (const float*)d_in[29];
  const float* L2W = (const float*)d_in[30]; const float* L2b = (const float*)d_in[31];
  const float* Wout = (const float*)d_in[32];
  /* d_in[33] = e1 (unused in reference forward) */
  const float* e2    = (const float*)d_in[34];
  const int*   msgI  = (const int*)d_in[35];
  const float* v2d   = (const float*)d_in[36];
  const int*   bei   = (const int*)d_in[37];
  const float* battr = (const float*)d_in[38];

  float* outv = (float*)d_out;              // [N]   final scalar head
  float* xout = (float*)d_out + NND;        // [N,64] GAT output (post-relu)

  // bump allocator on workspace
  char* wp = (char*)d_ws;
  auto alloc = [&](size_t bytes) -> void* {
    void* r = (void*)wp; wp += (bytes + 255) & ~(size_t)255; return r;
  };
  float*    v      = (float*)alloc((size_t)NND * 128 * 4);
  float*    h      = (float*)alloc((size_t)NND * 64 * 4);
  float*    x1     = (float*)alloc((size_t)NND * 64 * 4);
  float*    asrc   = (float*)alloc((size_t)NND * 4);
  float*    adst   = (float*)alloc((size_t)NND * 4);
  uint32_t* amax   = (uint32_t*)alloc((size_t)NND * 4);
  float*    den    = (float*)alloc((size_t)NND * 4);
  float*    aE     = (float*)alloc((size_t)EBN * 4);
  float*    exE    = (float*)alloc((size_t)EBN * 4);
  float*    we8    = (float*)alloc(256);
  float*    bcatf  = (float*)alloc((size_t)8192 * 192 * 4);
  uint16_t* bcatsw = (uint16_t*)alloc((size_t)256 * 12 * 512 * 2);
  float*    w2f    = (float*)alloc((size_t)8384 * 192 * 4);
  uint16_t* w2sw   = (uint16_t*)alloc((size_t)262 * 12 * 512 * 2);
  float*    we2c   = (float*)alloc((size_t)384 * 192 * 4);
  uint16_t* we2sw  = (uint16_t*)alloc((size_t)12 * 12 * 512 * 2);
  uint16_t* wh1sw  = (uint16_t*)alloc((size_t)4 * 8 * 512 * 2);
  uint16_t* wo1sw  = (uint16_t*)alloc((size_t)4 * 8 * 512 * 2);
  uint16_t* wh2sw  = (uint16_t*)alloc((size_t)2 * 4 * 512 * 2);
  uint16_t* wo2sw  = (uint16_t*)alloc((size_t)2 * 4 * 512 * 2);
  uint16_t* wdsw   = (uint16_t*)alloc((size_t)6 * 8 * 512 * 2);
  uint16_t* wusw   = (uint16_t*)alloc((size_t)4 * 16 * 512 * 2);
  uint16_t* l0sw   = (uint16_t*)alloc((size_t)8 * 16 * 512 * 2);
  uint16_t* l1sw   = (uint16_t*)alloc((size_t)8 * 16 * 512 * 2);
  uint16_t* l2sw   = (uint16_t*)alloc((size_t)8 * 16 * 512 * 2);
  float*    bzcat  = (float*)alloc(192 * 4);
  float*    be1e   = (float*)alloc(192 * 4);
  float*    be2e   = (float*)alloc(192 * 4);
  float*    zbuf   = (float*)alloc((size_t)NND * 192 * 4);
  float*    h1     = (float*)alloc((size_t)NND * 128 * 4);
  float*    g1b    = (float*)alloc((size_t)NND * 128 * 4);
  float*    o1     = (float*)alloc((size_t)NND * 128 * 4);
  float*    h2     = (float*)alloc((size_t)NND * 64 * 4);
  float*    g2b    = (float*)alloc((size_t)NND * 64 * 4);
  float*    o2     = (float*)alloc((size_t)NND * 64 * 4);
  float*    lin    = (float*)alloc((size_t)NND * 192 * 4);
  float*    o192   = (float*)alloc((size_t)NND * 192 * 4);
  float*    c384   = (float*)alloc((size_t)NND * 384 * 4);
  float*    v192   = (float*)alloc((size_t)NND * 192 * 4);
  float*    v128   = (float*)alloc((size_t)NND * 128 * 4);
  float*    va     = (float*)alloc((size_t)NND * 256 * 4);
  float*    vb     = (float*)alloc((size_t)NND * 256 * 4);

  dim3 b256(256);
  auto g1d = [](long total) { return dim3((unsigned)((total + 255) / 256)); };

  // ---- weight prep (bf16 + fragment swizzle) ----
  make_bcat_kernel<<<g1d(8192L * 192), b256, 0, stream>>>(B1, B2, bcatf);
  swizzle_kernel<<<g1d(256L * 12 * 512), b256, 0, stream>>>(bcatf, bcatsw, 8192, 256, 12);
  make_w2_kernel<<<g1d(8384L * 192), b256, 0, stream>>>(We1, w2f);
  swizzle_kernel<<<g1d(262L * 12 * 512), b256, 0, stream>>>(w2f, w2sw, 8384, 262, 12);
  make_we2c_kernel<<<g1d(384L * 192), b256, 0, stream>>>(We2, we2c);
  swizzle_kernel<<<g1d(12L * 12 * 512), b256, 0, stream>>>(we2c, we2sw, 384, 12, 12);
  swizzle_kernel<<<g1d(4L * 8 * 512), b256, 0, stream>>>(Wh1, wh1sw, 128, 4, 8);
  swizzle_kernel<<<g1d(4L * 8 * 512), b256, 0, stream>>>(Wo1, wo1sw, 128, 4, 8);
  swizzle_kernel<<<g1d(2L * 4 * 512), b256, 0, stream>>>(Wh2, wh2sw, 64, 2, 4);
  swizzle_kernel<<<g1d(2L * 4 * 512), b256, 0, stream>>>(Wo2, wo2sw, 64, 2, 4);
  swizzle_kernel<<<g1d(6L * 8 * 512), b256, 0, stream>>>(Wd, wdsw, 192, 6, 8);
  swizzle_kernel<<<g1d(4L * 16 * 512), b256, 0, stream>>>(Wu, wusw, 128, 4, 16);
  swizzle_kernel<<<g1d(8L * 16 * 512), b256, 0, stream>>>(L0W, l0sw, 256, 8, 16);
  swizzle_kernel<<<g1d(8L * 16 * 512), b256, 0, stream>>>(L1W, l1sw, 256, 8, 16);
  swizzle_kernel<<<g1d(8L * 16 * 512), b256, 0, stream>>>(L2W, l2sw, 256, 8, 16);
  bias_cat_kernel<<<1, 192, 0, stream>>>(bz1, 128, bz2, 64, bzcat);
  be1eff_kernel<<<1, 192, 0, stream>>>(be1, We1, be1e);
  be2eff_kernel<<<1, 192, 0, stream>>>(be2, We2, be2e);
  we8_kernel<<<1, 8, 0, stream>>>(gWe, gae, we8);

  // ---- v = segment_sum(e2, i) ----
  hipMemsetAsync(v, 0, (size_t)NND * 128 * 4, stream);
  scatter_rows_kernel<<<g1d((long)EMSG * 128), b256, 0, stream>>>(e2, msgI, v, EMSG);

  // ---- GAT layer 1 (v_2d -> x1) ----
  gat_node_kernel<<<NND, 64, 0, stream>>>(v2d, gW, gas, gad, h, asrc, adst);
  init_gat_out_kernel<<<g1d((long)NND * 64), b256, 0, stream>>>(x1, gbias, amax, den, NND);
  edge1_kernel<<<g1d(EBN), b256, 0, stream>>>(bei, battr, asrc, adst, we8, aE, amax, EBN);
  edge2_kernel<<<g1d(EBN), b256, 0, stream>>>(bei, aE, amax, exE, den, EBN);
  edge3_kernel<<<g1d((long)EBN * 64), b256, 0, stream>>>(bei, exE, den, h, x1, EBN);
  // ---- GAT layer 2 (x1 -> xout in d_out), then relu ----
  gat_node_kernel<<<NND, 64, 0, stream>>>(x1, gW, gas, gad, h, asrc, adst);
  init_gat_out_kernel<<<g1d((long)NND * 64), b256, 0, stream>>>(xout, gbias, amax, den, NND);
  edge1_kernel<<<g1d(EBN), b256, 0, stream>>>(bei, battr, asrc, adst, we8, aE, amax, EBN);
  edge2_kernel<<<g1d(EBN), b256, 0, stream>>>(bei, aE, amax, exE, den, EBN);
  edge3_kernel<<<g1d((long)EBN * 64), b256, 0, stream>>>(bei, exE, den, h, xout, EBN);
  relu_kernel<<<g1d((long)NND * 64), b256, 0, stream>>>(xout, (long)NND * 64);

  // ---- BilinearFusion + head (WMMA everywhere) ----
  dim3 gt((NND + 63) / 64), bt(128);
  dim3 gt2((NND + 63) / 64, 2), bt2(64);   // dual-A bilinear: 2 waves, N-split
  auto shg = [](int kch, int nt) {
    return (size_t)64 * (kch * 32 + 4) * 4 + (size_t)2 * nt * 1024;
  };

  gemm_bf16_kernel<8, 1><<<gt, bt, shg(4, 8), stream>>>(v, 128, 4, wh1sw, bh1, h1, NND);
  gemm_bf16_kernel<4, 1><<<gt, bt, shg(2, 4), stream>>>(xout, 64, 2, wh2sw, bh2, h2, NND);

  size_t sh1 = (size_t)(64 * 132 + 64 * 68) * 4 + (size_t)2 * 6 * 1024;
  bilinear1_kernel<<<gt2, bt2, sh1, stream>>>(v, xout, bcatsw, bzcat, zbuf, NND);

  gate_kernel<<<g1d((long)NND * 128), b256, 0, stream>>>(zbuf, 0, h1, g1b, NND, 128);
  gate_kernel<<<g1d((long)NND * 64), b256, 0, stream>>>(zbuf, 128, h2, g2b, NND, 64);
  gemm_bf16_kernel<8, 1><<<gt, bt, shg(4, 8), stream>>>(g1b, 128, 4, wo1sw, bo1, o1, NND);
  gemm_bf16_kernel<4, 1><<<gt, bt, shg(2, 4), stream>>>(g2b, 64, 2, wo2sw, bo2, o2, NND);

  concat2_kernel<<<g1d((long)NND * 192), b256, 0, stream>>>(o1, 128, o2, 64, lin, NND);
  size_t sh2 = (size_t)64 * 196 * 4 + (size_t)2 * 6 * 1024;
  bilinear2_kernel<<<gt2, bt2, sh2, stream>>>(lin, w2sw, be1e, o192, NND);

  concat3_kernel<<<g1d((long)NND * 384), b256, 0, stream>>>(o192, 192, o1, 128, o2, 64, c384, NND);
  gemm_bf16_kernel<12, 1><<<gt, bt, shg(12, 12), stream>>>(c384, 384, 12, we2sw, be2e, v192, NND);
  gemm_bf16_kernel<8, 0><<<gt, bt, shg(6, 8), stream>>>(v192, 192, 6, wdsw, bd, v128, NND);
  gemm_bf16_kernel<16, 0><<<gt, bt, shg(4, 16), stream>>>(v128, 128, 4, wusw, bu, va, NND);
  gemm_bf16_kernel<16, 2><<<gt, bt, shg(8, 16), stream>>>(va, 256, 8, l0sw, L0b, vb, NND);
  gemm_bf16_kernel<16, 2><<<gt, bt, shg(8, 16), stream>>>(vb, 256, 8, l1sw, L1b, va, NND);
  gemm_bf16_kernel<16, 2><<<gt, bt, shg(8, 16), stream>>>(va, 256, 8, l2sw, L2b, vb, NND);
  wout_kernel<<<g1d(NND), b256, 0, stream>>>(vb, Wout, outv, NND);
}
